// MoRModel_29016799052555
// MI455X (gfx1250) — compile-verified, matrix operand order checked by reading, not run on hardware
//
#include <hip/hip_runtime.h>
#include <stddef.h>
#include <stdint.h>

// ---------------- model dims ----------------
#define Bz   2
#define Sz   1024
#define Dz   1024
#define Hz   16
#define HDz  64
#define DFFz 4096
#define Lz   2
#define Rz   3
#define Vz   32000
#define Mz   (Bz*Sz)          // 2048 tokens

// ---------------- WMMA types ----------------
typedef __attribute__((ext_vector_type(16))) __bf16 bf16x16;
typedef __attribute__((ext_vector_type(8)))  float  floatx8;

union FragU { bf16x16 v; uint4 q[2]; };

__device__ inline bf16x16 ld_frag(const __bf16* base, int stride) {
    int lane = threadIdx.x & 31;
    int r  = lane & 15;           // row (A) / column (B)
    int kb = (lane >> 4) << 3;    // K sub-block select (0 or 8)
    const __bf16* p = base + r * stride + kb;
    FragU u;
    u.q[0] = *(const uint4*)p;          // K = kb .. kb+7
    u.q[1] = *(const uint4*)(p + 16);   // K = 16+kb .. 16+kb+7
    return u.v;
}

__device__ inline floatx8 wmma_bf16(bf16x16 a, bf16x16 b, floatx8 c) {
    return __builtin_amdgcn_wmma_f32_16x16x32_bf16(
        false, a, false, b, (short)0, c, false, false);
}

// ---- CDNA5 async global->LDS DMA (16B per lane), tracked by ASYNCcnt ----
__device__ inline void async_copy16(const void* gsrc, void* lds_dst) {
    uint32_t l = (uint32_t)(uintptr_t)lds_dst;   // generic LDS addr truncates to LDS offset
    uint64_t g = (uint64_t)(uintptr_t)gsrc;
    asm volatile("global_load_async_to_lds_b128 %0, %1, off"
                 :: "v"(l), "v"(g) : "memory");
}
__device__ inline void async_wait0() {
    asm volatile("s_wait_asynccnt 0x0" ::: "memory");
}

__device__ inline float hmax16(float v) {
    for (int m = 1; m < 16; m <<= 1) v = fmaxf(v, __shfl_xor(v, m, 32));
    return v;
}
__device__ inline float hsum16(float v) {
    for (int m = 1; m < 16; m <<= 1) v += __shfl_xor(v, m, 32);
    return v;
}

// ---------------- embed + router ----------------
__global__ __launch_bounds__(256)
void embed_router_kernel(const int* __restrict__ ids,
                         const float* __restrict__ emb,
                         const float* __restrict__ pos,
                         const float* __restrict__ Wr,
                         const float* __restrict__ br,
                         float* __restrict__ res,
                         float* __restrict__ depth_f,
                         int* __restrict__ depth_i,
                         float* __restrict__ acc) {
    int tok = blockIdx.x;                 // 0..2047
    int s   = tok & (Sz - 1);
    int id  = ids[tok];
    __shared__ float red[3 * 256];
    float p0 = 0.f, p1 = 0.f, p2 = 0.f;
    for (int d = threadIdx.x; d < Dz; d += 256) {
        float v = emb[(size_t)id * Dz + d] + pos[(size_t)s * Dz + d];
        res[(size_t)tok * Dz + d] = v;
        p0 += v * Wr[d * Rz + 0];
        p1 += v * Wr[d * Rz + 1];
        p2 += v * Wr[d * Rz + 2];
    }
    red[threadIdx.x] = p0; red[256 + threadIdx.x] = p1; red[512 + threadIdx.x] = p2;
    __syncthreads();
    for (int st = 128; st > 0; st >>= 1) {
        if ((int)threadIdx.x < st) {
            red[threadIdx.x]       += red[threadIdx.x + st];
            red[256 + threadIdx.x] += red[256 + threadIdx.x + st];
            red[512 + threadIdx.x] += red[512 + threadIdx.x + st];
        }
        __syncthreads();
    }
    if (threadIdx.x == 0) {
        float l0 = red[0] + br[0], l1 = red[256] + br[1], l2 = red[512] + br[2];
        float m  = fmaxf(l0, fmaxf(l1, l2));
        float e0 = __expf(l0 - m), e1 = __expf(l1 - m), e2 = __expf(l2 - m);
        float inv = 1.f / (e0 + e1 + e2);
        int dm = 1; float best = l0;
        if (l1 > best) { best = l1; dm = 2; }
        if (l2 > best) { best = l2; dm = 3; }
        depth_f[tok] = (float)dm;
        depth_i[tok] = dm;
        atomicAdd(&acc[0], e0 * inv);
        atomicAdd(&acc[1], e1 * inv);
        atomicAdd(&acc[2], e2 * inv);
    }
}

__global__ void zero_acc_kernel(float* acc) {
    if (threadIdx.x < 3) acc[threadIdx.x] = 0.f;
}

__global__ void loss_kernel(const float* __restrict__ acc, float* __restrict__ out) {
    if (threadIdx.x == 0) {
        float s = 0.f;
        for (int r = 0; r < Rz; ++r) {
            float imp = acc[r] / (float)Mz;
            s += imp * imp;
        }
        *out = (float)Rz * s;
    }
}

// ---------------- layernorm: fp32 row -> bf16 row ----------------
__global__ __launch_bounds__(256)
void ln_kernel(const float* __restrict__ x,
               const float* __restrict__ g,
               const float* __restrict__ b,
               __bf16* __restrict__ out) {
    int row = blockIdx.x;
    __shared__ float s1[256], s2[256];
    const float* xr = x + (size_t)row * Dz;
    float v[4], a = 0.f, sq = 0.f;
    for (int i = 0; i < 4; ++i) {
        v[i] = xr[threadIdx.x + i * 256];
        a += v[i]; sq += v[i] * v[i];
    }
    s1[threadIdx.x] = a; s2[threadIdx.x] = sq;
    __syncthreads();
    for (int st = 128; st > 0; st >>= 1) {
        if ((int)threadIdx.x < st) {
            s1[threadIdx.x] += s1[threadIdx.x + st];
            s2[threadIdx.x] += s2[threadIdx.x + st];
        }
        __syncthreads();
    }
    float mu  = s1[0] * (1.f / Dz);
    float var = s2[0] * (1.f / Dz) - mu * mu;
    float rstd = rsqrtf(var + 1e-5f);
    for (int i = 0; i < 4; ++i) {
        int d = threadIdx.x + i * 256;
        out[(size_t)row * Dz + d] = (__bf16)((v[i] - mu) * rstd * g[d] + b[d]);
    }
}

// ---------------- generic WMMA GEMM (128x128 tile, double-buffered) ----------
// C[M,N] = epilogue(A_bf16[M,K] @ W_f32 + bias), W indexed [k*N+n] (wt=0) or [n*K+k] (wt=1)
#define BMt 128
#define BNt 128
#define BKt 32
#define LDT 48   // LDS stride (elements); 96B rows keep uint4 alignment

__global__ __launch_bounds__(256, 1)
void gemm_kernel(const __bf16* __restrict__ A, const float* __restrict__ W,
                 const float* __restrict__ bias, const float* __restrict__ resid,
                 float* __restrict__ out_f, __bf16* __restrict__ out_b,
                 int M, int N, int K, int act, int wt) {
    __shared__ __bf16 As[2][BMt * LDT];
    __shared__ __bf16 Bs[2][BNt * LDT];
    int tid  = threadIdx.x;
    int wave = tid >> 5;
    int lane = tid & 31;
    int wr   = wave >> 1;            // 0..3  (row group, 32 rows each)
    int wc   = wave & 1;             // 0..1  (col group, 64 cols each)
    int rowb = blockIdx.y * BMt;
    int colb = blockIdx.x * BNt;

    floatx8 acc[2][4] = {};
    float4  breg[4];

    // per-thread A staging indices (2x 16B chunks of the 128x32 tile)
    int ar[2], akc[2];
    for (int it = 0; it < 2; ++it) {
        int t = tid + it * 256;
        ar[it]  = t >> 2;
        akc[it] = (t & 3) * 8;
    }

    // --- stage tile kt: A via async DMA straight to LDS[buf]; B into registers ---
    auto gload = [&](int kt, int buf) {
        int k0 = kt * BKt;
        for (int it = 0; it < 2; ++it)
            async_copy16(&A[(size_t)(rowb + ar[it]) * K + k0 + akc[it]],
                         &As[buf][ar[it] * LDT + akc[it]]);
        if (wt == 0) {
            for (int e = 0; e < 4; ++e) {
                int idx = tid + e * 256;          // 1024 float4 = 32k x 128n
                int k = idx >> 5, nq = idx & 31;
                breg[e] = *(const float4*)&W[(size_t)(k0 + k) * N + colb + nq * 4];
            }
        } else {
            for (int e = 0; e < 4; ++e) {
                int idx = tid + e * 256;          // 128n x 8 k-quads
                int n = idx >> 3, kq = idx & 7;
                breg[e] = *(const float4*)&W[(size_t)(colb + n) * K + k0 + kq * 4];
            }
        }
    };
    // --- convert+commit B registers into LDS buffer buf ---
    auto commitB = [&](int buf) {
        if (wt == 0) {
            for (int e = 0; e < 4; ++e) {
                int idx = tid + e * 256;
                int k = idx >> 5, nq = idx & 31;
                const float* f = (const float*)&breg[e];
                for (int j = 0; j < 4; ++j)
                    Bs[buf][(nq * 4 + j) * LDT + k] = (__bf16)f[j];
            }
        } else {
            for (int e = 0; e < 4; ++e) {
                int idx = tid + e * 256;
                int n = idx >> 3, kq = idx & 7;
                const float* f = (const float*)&breg[e];
                union { __bf16 h[4]; uint2 u; } t;
                for (int j = 0; j < 4; ++j) t.h[j] = (__bf16)f[j];
                *(uint2*)&Bs[buf][n * LDT + kq * 4] = t.u;   // 8B-aligned
            }
        }
    };

    const int KT = K / BKt;
    gload(0, 0);
    commitB(0);
    async_wait0();
    __syncthreads();

    for (int kt = 0; kt < KT; ++kt) {
        int cur = kt & 1;
        bool more = (kt + 1 < KT);
        if (more) gload(kt + 1, 1 - cur);        // async A + B loads overlap WMMAs

        // batch all fragment loads, then issue 8 back-to-back WMMAs
        bf16x16 af[2], bfr[4];
        const __bf16* abase = &As[cur][(wr * 32) * LDT];
        af[0] = ld_frag(abase,            LDT);
        af[1] = ld_frag(abase + 16 * LDT, LDT);
        for (int nt = 0; nt < 4; ++nt)
            bfr[nt] = ld_frag(&Bs[cur][(wc * 64 + nt * 16) * LDT], LDT);
        for (int mt = 0; mt < 2; ++mt)
            for (int nt = 0; nt < 4; ++nt)
                acc[mt][nt] = wmma_bf16(af[mt], bfr[nt], acc[mt][nt]);

        if (more) commitB(1 - cur);              // fill the other buffer
        async_wait0();                            // this wave's DMA landed
        __syncthreads();                          // publish across waves
    }

    int hi = lane >> 4, c0 = lane & 15;
    for (int mt = 0; mt < 2; ++mt) {
        for (int nt = 0; nt < 4; ++nt) {
            int c = colb + wc * 64 + nt * 16 + c0;
            float bv = bias ? bias[c] : 0.f;
            for (int i = 0; i < 8; ++i) {
                int r = rowb + wr * 32 + mt * 16 + i + 8 * hi;
                float v = acc[mt][nt][i] + bv;
                if (act == 1) {  // tanh-approx GELU (jax.nn.gelu default)
                    float u = v;
                    v = 0.5f * u * (1.f + tanhf(0.7978845608f * (u + 0.044715f * u * u * u)));
                }
                size_t o = (size_t)r * N + c;
                if (resid) v += resid[o];
                if (out_f) out_f[o] = v;
                if (out_b) out_b[o] = (__bf16)v;
            }
        }
    }
}

// ---------------- WMMA flash attention ----------------
// qkv: bf16 [M, 3*D]; O: bf16 [M, D]. Grid: (S/64, B*H), block 128 (4 waves).
__global__ __launch_bounds__(128)
void attn_kernel(const __bf16* __restrict__ qkv, __bf16* __restrict__ O) {
    __shared__ __bf16 Ks[32 * 72];        // [key][hd], stride 72 (144B)
    __shared__ __bf16 Vs[64 * 48];        // [hd][key], transposed, stride 48
    __shared__ __bf16 Ps[4][16 * 48];     // per-wave P tile [row][key]

    int tid = threadIdx.x, wave = tid >> 5, lane = tid & 31;
    int bh = blockIdx.y;
    int b  = bh / Hz, h = bh % Hz;
    int q0 = blockIdx.x * 64;
    int qb = q0 + wave * 16;
    int hi = lane >> 4, c0 = lane & 15;

    const __bf16* qbase = qkv + (size_t)(b * Sz + qb) * (3 * Dz) + h * HDz;
    bf16x16 qf0 = ld_frag(qbase,      3 * Dz);   // K dims 0..31
    bf16x16 qf1 = ld_frag(qbase + 32, 3 * Dz);   // K dims 32..63

    floatx8 accO[4] = {};
    float rm[8], rs[8];
    for (int i = 0; i < 8; ++i) { rm[i] = -1e30f; rs[i] = 0.f; }

    int nch = (q0 >> 5) + 2;            // key chunks of 32 covering 0..q0+63
    for (int ch = 0; ch < nch; ++ch) {
        int j0 = ch * 32;
        // K tile: async DMA (pure bf16 copy, row-major) -- 256 x 16B chunks
        for (int e = 0; e < 2; ++e) {
            int c = tid + e * 128;
            int key = c >> 3, h8 = (c & 7) * 8;
            async_copy16(qkv + (size_t)(b * Sz + j0 + key) * (3 * Dz) + h * HDz + Dz + h8,
                         &Ks[key * 72 + h8]);
        }
        // V tile: scalar transpose staging [hd][key]
        for (int e = 0; e < 16; ++e) {
            int idx = tid + e * 128;
            int key = idx >> 6, hd = idx & 63;
            Vs[hd * 48 + key] =
                qkv[(size_t)(b * Sz + j0 + key) * (3 * Dz) + h * HDz + 2 * Dz + hd];
        }
        async_wait0();
        __syncthreads();

        bool active = (j0 <= qb + 15);
        if (active) {
            // batch K-fragment loads, then the 4 score WMMAs
            bf16x16 kf[4];
            kf[0] = ld_frag(&Ks[0],            72);
            kf[1] = ld_frag(&Ks[32],           72);
            kf[2] = ld_frag(&Ks[16 * 72],      72);
            kf[3] = ld_frag(&Ks[16 * 72 + 32], 72);
            floatx8 s0 = {}, s1 = {};
            s0 = wmma_bf16(qf0, kf[0], s0);
            s1 = wmma_bf16(qf0, kf[2], s1);
            s0 = wmma_bf16(qf1, kf[1], s0);
            s1 = wmma_bf16(qf1, kf[3], s1);

            for (int i = 0; i < 8; ++i) {
                int q  = qb + i + 8 * hi;
                int ja = j0 + c0, jb = j0 + 16 + c0;
                s0[i] = (ja <= q) ? s0[i] * 0.125f : -1e30f;
                s1[i] = (jb <= q) ? s1[i] * 0.125f : -1e30f;
            }
            for (int i = 0; i < 8; ++i) {
                float m  = hmax16(fmaxf(s0[i], s1[i]));
                float nm = fmaxf(rm[i], m);
                float alpha = __expf(rm[i] - nm);
                rm[i] = nm;
                float p0 = __expf(s0[i] - nm);
                float p1 = __expf(s1[i] - nm);
                rs[i] = rs[i] * alpha + hsum16(p0 + p1);
                for (int nt = 0; nt < 4; ++nt) accO[nt][i] *= alpha;
                int row = i + 8 * hi;
                Ps[wave][row * 48 + c0]      = (__bf16)p0;
                Ps[wave][row * 48 + 16 + c0] = (__bf16)p1;
            }
        }
        __syncthreads();   // P tiles visible / ordered

        if (active) {
            bf16x16 pf = ld_frag(Ps[wave], 48);
            bf16x16 vf[4];
            for (int nt = 0; nt < 4; ++nt)
                vf[nt] = ld_frag(&Vs[nt * 16 * 48], 48);
            for (int nt = 0; nt < 4; ++nt)
                accO[nt] = wmma_bf16(pf, vf[nt], accO[nt]);
        }
        __syncthreads();   // before next chunk overwrites K/V
    }

    for (int nt = 0; nt < 4; ++nt) {
        for (int i = 0; i < 8; ++i) {
            int tok = b * Sz + qb + i + 8 * hi;
            int col = h * HDz + nt * 16 + c0;
            O[(size_t)tok * Dz + col] = (__bf16)(accO[nt][i] / rs[i]);
        }
    }
}

// ---------------- small elementwise kernels ----------------
__global__ void copy_kernel(const float* __restrict__ src, float* __restrict__ dst, int n) {
    int i = blockIdx.x * 256 + threadIdx.x;
    if (i < n) dst[i] = src[i];
}
__global__ void select_kernel(const int* __restrict__ depth,
                              const float* __restrict__ x,
                              float* __restrict__ res, int step) {
    int i = blockIdx.x * 256 + threadIdx.x;
    int row = i >> 10;                    // D = 1024
    if (depth[row] >= step + 1) res[i] = x[i];
}

// ---------------- host driver ----------------
static inline void launch_gemm(const __bf16* A, const float* W, const float* bias,
                               const float* resid, float* out_f, __bf16* out_b,
                               int M, int N, int K, int act, int wt, hipStream_t st) {
    dim3 grid(N / BNt, M / BMt);
    gemm_kernel<<<grid, 256, 0, st>>>(A, W, bias, resid, out_f, out_b, M, N, K, act, wt);
}

extern "C" void kernel_launch(void* const* d_in, const int* in_sizes, int n_in,
                              void* d_out, int out_size, void* d_ws, size_t ws_size,
                              hipStream_t stream) {
    (void)in_sizes; (void)n_in; (void)out_size; (void)ws_size;

    const int*   ids     = (const int*)  d_in[0];
    const float* emb     = (const float*)d_in[1];
    const float* pos     = (const float*)d_in[2];
    const float* Wqkv    = (const float*)d_in[3];
    const float* bqkv    = (const float*)d_in[4];
    const float* Wo      = (const float*)d_in[5];
    const float* bo      = (const float*)d_in[6];
    const float* W1      = (const float*)d_in[7];
    const float* b1      = (const float*)d_in[8];
    const float* W2      = (const float*)d_in[9];
    const float* b2      = (const float*)d_in[10];
    const float* ln1_g   = (const float*)d_in[11];
    const float* ln1_b   = (const float*)d_in[12];
    const float* ln2_g   = (const float*)d_in[13];
    const float* ln2_b   = (const float*)d_in[14];
    const float* final_g = (const float*)d_in[15];
    const float* final_b = (const float*)d_in[16];
    const float* Wr      = (const float*)d_in[17];
    const float* br      = (const float*)d_in[18];

    // workspace partition (256B-aligned chunks)
    char* ws = (char*)d_ws;
    size_t off = 0;
    auto take = [&](size_t bytes) { char* p = ws + off; off += (bytes + 255) & ~(size_t)255; return p; };
    float*  res    = (float*) take((size_t)Mz * Dz * 4);
    float*  x      = (float*) take((size_t)Mz * Dz * 4);
    __bf16* hbf    = (__bf16*)take((size_t)Mz * Dz * 2);
    __bf16* qkv    = (__bf16*)take((size_t)Mz * 3 * Dz * 2);
    __bf16* attnO  = (__bf16*)take((size_t)Mz * Dz * 2);
    __bf16* mlp1   = (__bf16*)take((size_t)Mz * DFFz * 2);
    int*    depth  = (int*)   take((size_t)Mz * 4);
    float*  racc   = (float*) take(256);

    float* logits  = (float*)d_out;                       // [M, V]
    float* depth_f = logits + (size_t)Mz * Vz;            // [M]
    float* loss    = depth_f + Mz;                        // [1]

    const int NE = Mz * Dz;

    // 1) router accumulator, embed + router
    zero_acc_kernel<<<1, 64, 0, stream>>>(racc);
    embed_router_kernel<<<Mz, 256, 0, stream>>>(ids, emb, pos, Wr, br,
                                                res, depth_f, depth, racc);
    loss_kernel<<<1, 64, 0, stream>>>(racc, loss);

    // 2) R recursion steps of the shared L-layer block
    for (int step = 0; step < Rz; ++step) {
        copy_kernel<<<NE / 256, 256, 0, stream>>>(res, x, NE);
        for (int l = 0; l < Lz; ++l) {
            const float* Wqkv_l = Wqkv + (size_t)l * Dz * 3 * Dz;
            const float* bqkv_l = bqkv + (size_t)l * 3 * Dz;
            const float* Wo_l   = Wo   + (size_t)l * Dz * Dz;
            const float* bo_l   = bo   + (size_t)l * Dz;
            const float* W1_l   = W1   + (size_t)l * Dz * DFFz;
            const float* b1_l   = b1   + (size_t)l * DFFz;
            const float* W2_l   = W2   + (size_t)l * DFFz * Dz;
            const float* b2_l   = b2   + (size_t)l * Dz;

            // attention sub-block
            ln_kernel<<<Mz, 256, 0, stream>>>(x, ln1_g + l * Dz, ln1_b + l * Dz, hbf);
            launch_gemm(hbf, Wqkv_l, bqkv_l, nullptr, nullptr, qkv,
                        Mz, 3 * Dz, Dz, 0, 0, stream);
            attn_kernel<<<dim3(Sz / 64, Bz * Hz), 128, 0, stream>>>(qkv, attnO);
            launch_gemm(attnO, Wo_l, bo_l, x, x, nullptr,
                        Mz, Dz, Dz, 0, 0, stream);

            // MLP sub-block
            ln_kernel<<<Mz, 256, 0, stream>>>(x, ln2_g + l * Dz, ln2_b + l * Dz, hbf);
            launch_gemm(hbf, W1_l, b1_l, nullptr, nullptr, mlp1,
                        Mz, DFFz, Dz, 1, 0, stream);
            launch_gemm(mlp1, W2_l, b2_l, x, x, nullptr,
                        Mz, Dz, DFFz, 0, 0, stream);
        }
        select_kernel<<<NE / 256, 256, 0, stream>>>(depth, x, res, step);
    }

    // 3) final LN + tied LM head (W = token_embed^T via wt=1)
    ln_kernel<<<Mz, 256, 0, stream>>>(res, final_g, final_b, hbf);
    launch_gemm(hbf, emb, nullptr, nullptr, logits, nullptr,
                Mz, Vz, Dz, 0, 1, stream);
}
